// CTLSTM_90598040141950
// MI455X (gfx1250) — compile-verified, matrix operand order checked by reading, not run on hardware
//
#include <hip/hip_runtime.h>
#include <hip/hip_bf16.h>

#define T_STEPS 4096
#define H       1024
#define G7      7168            // 7*H
#define NTILES  (G7 / 16)       // 448 col tiles
#define KBLK    (H / 32)        // 32 k-blocks of 32
#define NBLK    56              // persistent blocks for recurrent kernel
#define KSLICES 4
#define KS      (H / KSLICES)   // 256 k per slice
#define CGROUPS (NBLK / KSLICES)// 14 column groups
#define PAIRS   (G7 / 2)        // 3584 bf16 column-pairs
#define PPG     (PAIRS / CGROUPS) // 256 pairs per group (== blockDim)

typedef __attribute__((ext_vector_type(16))) __bf16 v16bf;
typedef __attribute__((ext_vector_type(8)))  float  v8f;

union Frag {                     // 32B fragment: two b128 loads <-> one WMMA operand
    uint4 u[2];
    v16bf v;
};
struct Frags { Frag a, b0, b1, b2, b3; };

__device__ __forceinline__ unsigned short f2bf(float f) {
    unsigned u = __float_as_uint(f);
    u += 0x7FFFu + ((u >> 16) & 1u);      // round to nearest even
    return (unsigned short)(u >> 16);
}
__device__ __forceinline__ float sigmoidf_(float x) { return 1.0f / (1.0f + __expf(-x)); }

__device__ __forceinline__ void load_frags(Frags& f,
                                           const unsigned short* __restrict__ av,
                                           const unsigned short* __restrict__ bv) {
    f.a.u[0]  = *(const uint4*)(av);                   f.a.u[1]  = *(const uint4*)(av + 8);
    f.b0.u[0] = *(const uint4*)(bv);                   f.b0.u[1] = *(const uint4*)(bv + 8);
    f.b1.u[0] = *(const uint4*)(bv + 1 * KBLK * 512);  f.b1.u[1] = *(const uint4*)(bv + 1 * KBLK * 512 + 8);
    f.b2.u[0] = *(const uint4*)(bv + 2 * KBLK * 512);  f.b2.u[1] = *(const uint4*)(bv + 2 * KBLK * 512 + 8);
    f.b3.u[0] = *(const uint4*)(bv + 3 * KBLK * 512);  f.b3.u[1] = *(const uint4*)(bv + 3 * KBLK * 512 + 8);
}

// ---------------------------------------------------------------- phase 0a: W2 (recurrent half) fp32 -> bf16, K-major
__global__ void __launch_bounds__(256) convert_w2(const float* __restrict__ w_r,
                                                  unsigned short* __restrict__ w2) {
    long i = (long)blockIdx.x * 256 + threadIdx.x;
    if (i < (long)H * G7)
        w2[i] = f2bf(w_r[(long)H * G7 + i]);       // rows H..2H-1
}

// ---------------------------------------------------------------- phase 0b: W1 (emb half) -> WMMA-fragment-packed bf16
// Bpk[nt][kk32][lane][i]: wave's B fragment = one contiguous 1KB line (lane: 2 x b128).
// ISA 7.12.2: lane-half kb = (lane<16)?0:8; fragment K = kb+{0..7} then kb+{16..23}.
__global__ void __launch_bounds__(256) pack_w1(const float* __restrict__ w_r,
                                               unsigned short* __restrict__ Bpk) {
    long idx = (long)blockIdx.x * 256 + threadIdx.x;
    if (idx >= (long)H * G7) return;
    int i    = (int)(idx & 15);
    int lane = (int)((idx >> 4) & 31);
    int kk32 = (int)((idx >> 9) & 31);
    int nt   = (int)(idx >> 14);
    int kb   = (lane < 16) ? 0 : 8;
    int k    = kk32 * 32 + kb + ((i < 8) ? i : (8 + i));
    int col  = nt * 16 + (lane & 15);
    Bpk[idx] = f2bf(w_r[(long)k * G7 + col]);
}

// ---------------------------------------------------------------- phase 0c: gather emb_event[event] -> fragment-packed bf16
__global__ void __launch_bounds__(256) gather_emb(const int* __restrict__ event,
                                                  const float* __restrict__ emb_event,
                                                  unsigned short* __restrict__ Apk) {
    int t = blockIdx.x;
    int mtile = t >> 4, m = t & 15;
    const float* src = emb_event + (long)event[t] * H;
    for (int j = threadIdx.x; j < H; j += blockDim.x) {
        int kk32 = j >> 5, koff = j & 31;
        int lane = m + (((koff >> 3) & 1) << 4);
        int i    = (koff & 7) + ((koff & 16) ? 8 : 0);
        long pidx = (((long)mtile * KBLK + kk32) * 32 + lane) * 16 + i;
        Apk[pidx] = f2bf(src[j]);
    }
}

// ---------------------------------------------------------------- phase 1: pre[T,7H] = emb @ W1 + b  (WMMA bf16)
// Wave -> 16x64 strip, 4 accumulators. Register double-buffered fragments:
// iteration k+1's 10 b128 loads are in flight while iteration k's 4 WMMAs execute.
__global__ void __launch_bounds__(128) gemm_pre(const unsigned short* __restrict__ Apk,
                                                const unsigned short* __restrict__ Bpk,
                                                const float* __restrict__ bias,
                                                float* __restrict__ pre) {
    const int lane   = threadIdx.x & 31;
    const int wave   = threadIdx.x >> 5;
    const int ntbase = blockIdx.x * 4;              // 4 col tiles per wave
    const int mtile  = blockIdx.y * 4 + wave;

    const unsigned short* ap = Apk + ((long)mtile  * KBLK) * 512 + lane * 16;
    const unsigned short* bp = Bpk + ((long)ntbase * KBLK) * 512 + lane * 16;

    v8f acc0 = {}, acc1 = {}, acc2 = {}, acc3 = {};
    Frags f[2];
    load_frags(f[0], ap, bp);                       // prologue
#pragma unroll 2
    for (int kk32 = 0; kk32 < KBLK; kk32++) {
        const int cur = kk32 & 1;
        if (kk32 + 1 < KBLK)                        // prefetch next k-block into other buffer
            load_frags(f[cur ^ 1], ap + (long)(kk32 + 1) * 512, bp + (long)(kk32 + 1) * 512);
        acc0 = __builtin_amdgcn_wmma_f32_16x16x32_bf16(false, f[cur].a.v, false, f[cur].b0.v, (short)0, acc0, false, false);
        acc1 = __builtin_amdgcn_wmma_f32_16x16x32_bf16(false, f[cur].a.v, false, f[cur].b1.v, (short)0, acc1, false, false);
        acc2 = __builtin_amdgcn_wmma_f32_16x16x32_bf16(false, f[cur].a.v, false, f[cur].b2.v, (short)0, acc2, false, false);
        acc3 = __builtin_amdgcn_wmma_f32_16x16x32_bf16(false, f[cur].a.v, false, f[cur].b3.v, (short)0, acc3, false, false);
    }

    const int mr = mtile * 16 + ((lane >> 4) << 3); // C/D layout: VGPR r -> M=r (+8 upper lane-half)
    const int nc = lane & 15;
#pragma unroll
    for (int q = 0; q < 4; q++) {
        int ncol = (ntbase + q) * 16 + nc;
        float bs = bias[ncol];
        v8f acc = (q == 0) ? acc0 : (q == 1) ? acc1 : (q == 2) ? acc2 : acc3;
#pragma unroll
        for (int r = 0; r < 8; r++)
            pre[(long)(mr + r) * G7 + ncol] = acc[r] + bs;
    }
}

// ---------------------------------------------------------------- phase 2a: init triple-buffered gates + state + barrier
__global__ void __launch_bounds__(256) init_state(const float* __restrict__ pre,
                                                  float* __restrict__ gbuf,   // [3][7H]
                                                  float* __restrict__ h_d,    // [H]
                                                  unsigned* __restrict__ sync) {
    int g = blockIdx.x * 256 + threadIdx.x;
    if (g < G7) {
        gbuf[g]          = pre[g];
        gbuf[G7 + g]     = pre[G7 + g];
        gbuf[2 * G7 + g] = pre[2 * G7 + g];
    }
    if (g < H) h_d[g] = 0.0f;
    if (g < 2) sync[g] = 0u;
}

// ---------------------------------------------------------------- grid-wide sense barrier (persistent kernel)
__device__ __forceinline__ void grid_sync(unsigned* cnt, unsigned* gen) {
    __threadfence();
    __syncthreads();
    if (threadIdx.x == 0) {
        unsigned g = __atomic_load_n(gen, __ATOMIC_ACQUIRE);
        if (atomicAdd(cnt, 1u) == NBLK - 1u) {
            atomicExch(cnt, 0u);
            __threadfence();
            atomicAdd(gen, 1u);
        } else {
            while (__atomic_load_n(gen, __ATOMIC_ACQUIRE) == g)
                __builtin_amdgcn_s_sleep(1);
        }
    }
    __syncthreads();
}

// ---------------------------------------------------------------- phase 2b: sequential CT-LSTM scan
__global__ void __launch_bounds__(256) recurrent(const float* __restrict__ pre,
                                                 const unsigned* __restrict__ W2pk,  // [H][PAIRS] packed 2xbf16
                                                 const float* __restrict__ duration,
                                                 float* __restrict__ gbuf,
                                                 float* __restrict__ h_d,
                                                 unsigned* __restrict__ sync,
                                                 float* __restrict__ out) {
    __shared__ float hsh[KS];
    const int tid    = threadIdx.x;
    const int blk    = blockIdx.x;
    const int kslice = blk % KSLICES;
    const int cgrp   = blk / KSLICES;
    const int pair   = cgrp * PPG + tid;        // owns columns 2*pair, 2*pair+1
    const int g      = blk * 256 + tid;         // global thread id (fixed ownership)

    float c_reg = 0.0f, cb_reg = 0.0f;          // persistent cell states (threads g < H)

    for (int t = 0; t < T_STEPS; t++) {
        float* cur = gbuf + (size_t)(t % 3) * G7;

        // ---- phase A: recurrent matvec partial (k-slice x column-pair)
        hsh[tid] = h_d[kslice * KS + tid];
        __syncthreads();
        float a0 = 0.0f, a1 = 0.0f;
        const unsigned* wp = W2pk + (long)(kslice * KS) * PAIRS + pair;
#pragma unroll 8
        for (int k = 0; k < KS; k++) {
            unsigned w  = wp[(long)k * PAIRS];            // 2 bf16 columns, coalesced
            float    hv = hsh[k];
            a0 = fmaf(hv, __uint_as_float(w << 16), a0);
            a1 = fmaf(hv, __uint_as_float(w & 0xFFFF0000u), a1);
        }
        atomicAdd(&cur[2 * pair],     a0);
        atomicAdd(&cur[2 * pair + 1], a1);
        grid_sync(sync, sync + 1);

        // ---- phase B: elementwise gating (threads g < H own hidden unit j = g)
        if (g < H) {
            int   j   = g;
            float gi  = sigmoidf_(cur[j]);
            float gf  = sigmoidf_(cur[H + j]);
            float gz  = tanhf(cur[2 * H + j]);
            float go  = sigmoidf_(cur[3 * H + j]);
            float gib = sigmoidf_(cur[4 * H + j]);
            float gfb = sigmoidf_(cur[5 * H + j]);
            float x   = cur[6 * H + j];
            float gdl = (x > 20.0f) ? x : log1pf(__expf(x));   // softplus

            float ct  = gf * c_reg + gi * gz;
            float cbt = gfb * cb_reg + gib * gz;
            float ht  = go * tanhf(ct);
            float dtv = duration[t];
            float cd  = cbt + (ct - cbt) * __expf(-gdl * dtv);
            c_reg = ct; cb_reg = cbt;
            h_d[j] = go * tanhf(cd);

            long base = (long)t * H + j;
            out[base]                         = ht;
            out[(long)1 * T_STEPS * H + base] = ct;
            out[(long)2 * T_STEPS * H + base] = cbt;
            out[(long)3 * T_STEPS * H + base] = go;
            out[(long)4 * T_STEPS * H + base] = gdl;
        }
        // reset the buffer step t+2 will accumulate into (last read at step t-1; safe)
        if (g < G7 && (t + 2) < T_STEPS)
            gbuf[(size_t)((t + 2) % 3) * G7 + g] = pre[(long)(t + 2) * G7 + g];

        grid_sync(sync, sync + 1);
    }
}

// ---------------------------------------------------------------- host-side launch
extern "C" void kernel_launch(void* const* d_in, const int* in_sizes, int n_in,
                              void* d_out, int out_size, void* d_ws, size_t ws_size,
                              hipStream_t stream) {
    (void)in_sizes; (void)n_in; (void)out_size; (void)ws_size;
    const int*   event     = (const int*)d_in[0];
    const float* duration  = (const float*)d_in[1];
    const float* w_r       = (const float*)d_in[2];
    const float* b_r       = (const float*)d_in[3];
    const float* emb_event = (const float*)d_in[4];
    float*       out       = (float*)d_out;

    char*  ws = (char*)d_ws;
    size_t o  = 0;
    auto alloc = [&](size_t bytes) -> void* {
        o = (o + 255) & ~(size_t)255;
        void* p = ws + o;
        o += bytes;
        return p;
    };
    unsigned short* w1pk  = (unsigned short*)alloc((size_t)H * G7 * 2);        // 14.7 MB (fragment-packed)
    unsigned short* w2    = (unsigned short*)alloc((size_t)H * G7 * 2);        // 14.7 MB (K-major)
    unsigned short* apk   = (unsigned short*)alloc((size_t)T_STEPS * H * 2);   //  8.4 MB (fragment-packed)
    float*          pre   = (float*)alloc((size_t)T_STEPS * G7 * 4);           // 117 MB
    float*          gbuf  = (float*)alloc((size_t)3 * G7 * 4);
    float*          h_d   = (float*)alloc((size_t)H * 4);
    unsigned*       syncb = (unsigned*)alloc(2 * sizeof(unsigned));

    long nW = (long)H * G7;
    convert_w2<<<dim3((unsigned)((nW + 255) / 256)), dim3(256), 0, stream>>>(w_r, w2);
    pack_w1<<<dim3((unsigned)((nW + 255) / 256)), dim3(256), 0, stream>>>(w_r, w1pk);
    gather_emb<<<dim3(T_STEPS), dim3(256), 0, stream>>>(event, emb_event, apk);
    gemm_pre<<<dim3(NTILES / 4, T_STEPS / 16 / 4), dim3(128), 0, stream>>>(apk, w1pk, b_r, pre);
    init_state<<<dim3((G7 + 255) / 256), dim3(256), 0, stream>>>(pre, gbuf, h_d, syncb);
    recurrent<<<dim3(NBLK), dim3(256), 0, stream>>>(pre, (const unsigned*)w2, duration,
                                                    gbuf, h_d, syncb, out);
}
